// SO2_EquiConv_71554155151872
// MI455X (gfx1250) — matrix operand order repaired and checked by reading, not compiled, not run on hardware
//
#include <hip/hip_runtime.h>
#include <hip/hip_bf16.h>

typedef __attribute__((ext_vector_type(8)))  __bf16 v8bf;
typedef __attribute__((ext_vector_type(16))) __bf16 v16bf;
typedef __attribute__((ext_vector_type(8)))  float  v8f;

#define WAVES 4
#define SX_STRIDE 808            // 800 features + pad (1616B rows -> conflict-free frag loads)
#define WIG_STRIDE 40            // 32 used + pad (80B rows)
#define YT_CHAN_STRIDE 40
#define YT_EDGE_STRIDE (32 * YT_CHAN_STRIDE)

// bf16-element offsets of (transposed, bf16) weights in workspace
constexpr size_t OFF_W0  = 0;        // [160][160]
constexpr size_t OFF_W1R = 25600;    // [128][128]
constexpr size_t OFF_W1I = 41984;
constexpr size_t OFF_W2R = 58368;    // [96][96]
constexpr size_t OFF_W2I = 67584;
constexpr size_t OFF_W3R = 76800;    // [64][64]
constexpr size_t OFF_W3I = 80896;
constexpr size_t OFF_W4R = 84992;    // [32][32]
constexpr size_t OFF_W4I = 86016;
constexpr size_t OFF_FCW = 87040;    // [160][64]

// coeff p (= l^2+l+m) -> base feature offset in the batched SO(2) layout
__constant__ int c_featOff[25] = {
  0, 288, 32, 160, 512, 320, 64, 192, 416, 672, 544, 352, 96,
  224, 448, 608, 768, 704, 576, 384, 128, 256, 480, 640, 736};
// feature block q (= f/32) -> coeff p
__constant__ int c_invFeat[25] = {
  0, 2, 6, 12, 20,  3, 7, 13, 21,  1, 5, 11, 19,
  8, 14, 22,  4, 10, 18,  15, 23,  9, 17,  24, 16};

__device__ __forceinline__ void lds_fence() {
  asm volatile("s_wait_dscnt 0" ::: "memory");
}

__device__ __forceinline__ v8f bwmma(v16bf a, v16bf b, v8f c) {
  return __builtin_amdgcn_wmma_f32_16x16x32_bf16(false, a, false, b, (short)0, c,
                                                 false, false);
}

// A fragment (16x32 bf16): lanes 0-15 rows 0-15 with K {0..7,16..23},
// lanes 16-31 same rows with K {8..15,24..31}; base points at [row0][k0].
__device__ __forceinline__ v16bf load_frag_A(const __bf16* base, int stride, int lane) {
  const int half = lane >> 4, r = lane & 15;
  const __bf16* p = base + r * stride + half * 8;
  v8bf lo = *(const v8bf*)p;
  v8bf hi = *(const v8bf*)(p + 16);
  return __builtin_shufflevector(lo, hi, 0, 1, 2, 3, 4, 5, 6, 7, 8, 9, 10, 11, 12, 13, 14, 15);
}

// B fragment (32x16 bf16) from B^T stored row-major [N][K]:
// lanes 0-15: col n=lane, K {0..7,16..23}; lanes 16-31: K {8..15,24..31}.
__device__ __forceinline__ v16bf load_frag_B(const __bf16* baseT, int stride, int lane) {
  const int half = lane >> 4, n = lane & 15;
  const __bf16* p = baseT + n * stride + half * 8;
  v8bf lo = *(const v8bf*)p;
  v8bf hi = *(const v8bf*)(p + 16);
  return __builtin_shufflevector(lo, hi, 0, 1, 2, 3, 4, 5, 6, 7, 8, 9, 10, 11, 12, 13, 14, 15);
}

// Store one scaled Y tile (rows=edges, cols=features) into per-edge transposed
// staging buffer sYT[edge][chan][coeff] for the rotate-back B operand.
__device__ __forceinline__ void store_y(__bf16* sYT, v8f y, int fbase, int lane) {
  const int half = lane >> 4, col = lane & 15;
  const int f = fbase + col;
  const int coeff = c_invFeat[f >> 5];
  const int chan = f & 31;
  __bf16* p = sYT + chan * YT_CHAN_STRIDE + coeff;
#pragma unroll
  for (int v = 0; v < 8; ++v)
    p[(half * 8 + v) * YT_EDGE_STRIDE] = (__bf16)y[v];
}

// SO(2) block for m=M:  Yp = Xp*Wr - Xn*Wi ; Yn = Xp*Wi + Xn*Wr ; then scale.
template <int M>
__device__ __forceinline__ void so2_block(const __bf16* sX, __bf16* sYT,
                                          const __bf16* wsb, const v8f (&sc)[10],
                                          int lane) {
  constexpr int Dm = (5 - M) * 32;
  constexpr int NK = Dm / 32;
  constexpr int NJ = Dm / 16;
  constexpr int PO[5] = {0, 160, 416, 608, 736};
  constexpr int NO[5] = {0, 288, 512, 672, 768};
  constexpr size_t WR[5] = {0, OFF_W1R, OFF_W2R, OFF_W3R, OFF_W4R};
  constexpr size_t WI[5] = {0, OFF_W1I, OFF_W2I, OFF_W3I, OFF_W4I};
  const __bf16* wr = wsb + WR[M];
  const __bf16* wi = wsb + WI[M];
#pragma unroll
  for (int j = 0; j < NJ; ++j) {
    v8f a1 = {}, a2 = {};
#pragma unroll
    for (int k = 0; k < NK; ++k) {
      a1 = bwmma(load_frag_A(sX + PO[M] + k * 32, SX_STRIDE, lane),
                 load_frag_B(wr + (size_t)(16 * j) * Dm + k * 32, Dm, lane), a1);
      a2 = bwmma(load_frag_A(sX + NO[M] + k * 32, SX_STRIDE, lane),
                 load_frag_B(wi + (size_t)(16 * j) * Dm + k * 32, Dm, lane), a2);
    }
    v8f s = sc[2 * M + j];              // scale tile aligns lane-for-lane with Y tile
    store_y(sYT, (a1 - a2) * s, PO[M] + 16 * j, lane);
    v8f a3 = {};
#pragma unroll
    for (int k = 0; k < NK; ++k) {
      a3 = bwmma(load_frag_A(sX + PO[M] + k * 32, SX_STRIDE, lane),
                 load_frag_B(wi + (size_t)(16 * j) * Dm + k * 32, Dm, lane), a3);
      a3 = bwmma(load_frag_A(sX + NO[M] + k * 32, SX_STRIDE, lane),
                 load_frag_B(wr + (size_t)(16 * j) * Dm + k * 32, Dm, lane), a3);
    }
    store_y(sYT, a3 * s, NO[M] + 16 * j, lane);
  }
}

// Weight prep: f32 [K][N] -> bf16 transposed [N][K] in workspace (L2-resident).
__global__ void wprep_kernel(const float* __restrict__ src, __bf16* __restrict__ dst,
                             int K, int N) {
  int i = blockIdx.x * blockDim.x + threadIdx.x;
  if (i >= K * N) return;
  int k = i / N, n = i - k * N;
  dst[(size_t)n * K + k] = (__bf16)src[i];
}

__global__ void __launch_bounds__(WAVES * 32, 1)
so2_equiconv_main(const float* __restrict__ x, const float* __restrict__ wig,
                  const float* __restrict__ efc, const int* __restrict__ esrc,
                  const int* __restrict__ edst, const float* __restrict__ fcb,
                  const __bf16* __restrict__ wsb, float* __restrict__ out, int NT) {
  __shared__ __align__(16) __bf16 sWigA[WAVES][32 * WIG_STRIDE];
  __shared__ __align__(16) __bf16 sXBA[WAVES][32 * WIG_STRIDE];
  __shared__ __align__(16) __bf16 sXA[WAVES][16 * SX_STRIDE];
  __shared__ __align__(16) __bf16 sYTA[WAVES][16 * YT_EDGE_STRIDE];

  const int lane = threadIdx.x & 31;
  const int wv = threadIdx.x >> 5;
  __bf16* sWig = sWigA[wv];
  __bf16* sXB = sXBA[wv];
  __bf16* sX = sXA[wv];
  __bf16* sYT = sYTA[wv];
  const int half = lane >> 4, col = lane & 15;
  const int lastE = NT * 16 - 1;

  // Hoisted per-lane feature-offset tables for the rotation scatter.
  // ci = half*8+v is always < 25; only the +16 side needs a mask.
  int foA[8], foB[8];
  bool g2[8];
#pragma unroll
  for (int v = 0; v < 8; ++v) {
    foA[v] = c_featOff[half * 8 + v];
    const int ci2 = half * 8 + v + 16;
    g2[v] = ci2 < 25;
    foB[v] = g2[v] ? c_featOff[ci2] : 0;
  }

  for (int T = blockIdx.x * WAVES + wv; T < NT; T += gridDim.x * WAVES) {
    // ---- Phase 1: gather + Wigner rotation, per edge 32x32x32 bf16 WMMA ----
    for (int t = 0; t < 16; ++t) {
      const int e = (T << 4) + t;
      const int src = esrc[e];
      // Prefetch next edge's wigner tile + gathered x row (global_prefetch_b8).
      {
        const int e1 = e < lastE ? e + 1 : e;
        const int src1 = esrc[e1];
        __builtin_prefetch(wig + (size_t)e1 * 625 + lane * 20, 0, 3);
        __builtin_prefetch(x + (size_t)src1 * 800 + lane * 25, 0, 3);
      }
      lds_fence();
      const float* wge = wig + (size_t)e * 625;
      const float* xe = x + (size_t)src * 800;
      for (int idx = lane; idx < 1024; idx += 32) {
        const int i = idx >> 5, j = idx & 31;  // j = lane: contiguous global reads
        sWig[i * WIG_STRIDE + j] = (__bf16)((i < 25 && j < 25) ? wge[i * 25 + j] : 0.f);
        // B^T operand: sXB[chan][coeff] = x[coeff][chan]; transpose on LDS store
        sXB[j * WIG_STRIDE + i] = (__bf16)((i < 25) ? xe[i * 32 + j] : 0.f);
      }
      lds_fence();
      v16bf a0 = load_frag_A(sWig, WIG_STRIDE, lane);
      v16bf a1 = load_frag_A(sWig + 16 * WIG_STRIDE, WIG_STRIDE, lane);
      v16bf b0 = load_frag_B(sXB, WIG_STRIDE, lane);
      v16bf b1 = load_frag_B(sXB + 16 * WIG_STRIDE, WIG_STRIDE, lane);
      v8f z = {};
      v8f c00 = bwmma(a0, b0, z), c01 = bwmma(a0, b1, z);
      v8f c10 = bwmma(a1, b0, z), c11 = bwmma(a1, b1, z);
      // scatter xr into batched SO(2) layout sX[t][featOff[coeff]+chan]
      __bf16* xrow = sX + t * SX_STRIDE;
#pragma unroll
      for (int v = 0; v < 8; ++v) {
        xrow[foA[v] + col] = (__bf16)c00[v];
        xrow[foA[v] + 16 + col] = (__bf16)c01[v];
        if (g2[v]) {
          xrow[foB[v] + col] = (__bf16)c10[v];
          xrow[foB[v] + 16 + col] = (__bf16)c11[v];
        }
      }
    }
    lds_fence();

    // ---- Phase 2: scale = silu(edge_fc @ fc_w + b), kept as 10 C-fragments ----
    v8f sc[10];
    {
      const float* fbase = efc + (size_t)((T << 4) + col) * 64 + half * 8;
#pragma unroll
      for (int j = 0; j < 10; ++j) {
        v8f acc = {};
#pragma unroll
        for (int k = 0; k < 2; ++k) {
          const float* fp = fbase + k * 32;
          v16bf a;
#pragma unroll
          for (int q = 0; q < 8; ++q) {
            a[q] = (__bf16)fp[q];
            a[q + 8] = (__bf16)fp[q + 16];
          }
          v16bf b = load_frag_B(wsb + OFF_FCW + (size_t)(16 * j) * 64 + k * 32, 64, lane);
          acc = bwmma(a, b, acc);
        }
        const float bias = fcb[16 * j + col];
#pragma unroll
        for (int v = 0; v < 8; ++v) {
          float w = acc[v] + bias;
          acc[v] = w / (1.f + __expf(-w));
        }
        sc[j] = acc;
      }
    }

    // ---- Phase 3: SO(2) batched GEMMs (M = 16 edges), scale, stage into sYT ----
#pragma unroll
    for (int j = 0; j < 10; ++j) {   // m = 0 block: [16,160] @ [160,160]
      v8f acc = {};
#pragma unroll
      for (int k = 0; k < 5; ++k) {
        acc = bwmma(load_frag_A(sX + k * 32, SX_STRIDE, lane),
                    load_frag_B(wsb + OFF_W0 + (size_t)(16 * j) * 160 + k * 32, 160, lane),
                    acc);
      }
      store_y(sYT, acc * sc[j], 16 * j, lane);
    }
    so2_block<1>(sX, sYT, wsb, sc, lane);
    so2_block<2>(sX, sYT, wsb, sc, lane);
    so2_block<3>(sX, sYT, wsb, sc, lane);
    so2_block<4>(sX, sYT, wsb, sc, lane);

    // ---- Phase 4: rotate back with W^T, scatter-add to destination nodes ----
    for (int t = 0; t < 16; ++t) {
      const int e = (T << 4) + t;
      {
        const int e1 = e < lastE ? e + 1 : e;
        __builtin_prefetch(wig + (size_t)e1 * 625 + lane * 20, 0, 3);
      }
      lds_fence();
      const float* wge = wig + (size_t)e * 625;
      for (int idx = lane; idx < 1024; idx += 32) {
        const int i = idx >> 5, j = idx & 31;  // read W row-major (coalesced),
        // store transposed so sWig[j][i] = W[i][j]  =>  sWig = W^T (A operand)
        const float wvv = (i < 25 && j < 25) ? wge[i * 25 + j] : 0.f;
        sWig[j * WIG_STRIDE + i] = (__bf16)wvv;
      }
      lds_fence();
      const __bf16* yb = sYT + t * YT_EDGE_STRIDE;
      v16bf a0 = load_frag_A(sWig, WIG_STRIDE, lane);
      v16bf a1 = load_frag_A(sWig + 16 * WIG_STRIDE, WIG_STRIDE, lane);
      v16bf b0 = load_frag_B(yb, YT_CHAN_STRIDE, lane);
      v16bf b1 = load_frag_B(yb + 16 * YT_CHAN_STRIDE, YT_CHAN_STRIDE, lane);
      v8f z = {};
      v8f c00 = bwmma(a0, b0, z), c01 = bwmma(a0, b1, z);
      v8f c10 = bwmma(a1, b0, z), c11 = bwmma(a1, b1, z);
      float* ob = out + (size_t)edst[e] * 800;
#pragma unroll
      for (int v = 0; v < 8; ++v) {
        const int i0 = half * 8 + v;            // always < 25
        unsafeAtomicAdd(ob + i0 * 32 + col, c00[v]);
        unsafeAtomicAdd(ob + i0 * 32 + 16 + col, c01[v]);
        if (g2[v]) {                            // i0 + 16 < 25
          unsafeAtomicAdd(ob + (i0 + 16) * 32 + col, c10[v]);
          unsafeAtomicAdd(ob + (i0 + 16) * 32 + 16 + col, c11[v]);
        }
      }
    }
  }
}

extern "C" void kernel_launch(void* const* d_in, const int* in_sizes, int n_in,
                              void* d_out, int out_size, void* d_ws, size_t ws_size,
                              hipStream_t stream) {
  const float* x = (const float*)d_in[0];
  const float* wig = (const float*)d_in[1];
  const float* efc = (const float*)d_in[2];
  const int* esrc = (const int*)d_in[3];
  const int* edst = (const int*)d_in[4];
  const float* w0 = (const float*)d_in[5];
  const float* w1r = (const float*)d_in[6];
  const float* w1i = (const float*)d_in[7];
  const float* w2r = (const float*)d_in[8];
  const float* w2i = (const float*)d_in[9];
  const float* w3r = (const float*)d_in[10];
  const float* w3i = (const float*)d_in[11];
  const float* w4r = (const float*)d_in[12];
  const float* w4i = (const float*)d_in[13];
  const float* fcw = (const float*)d_in[14];
  const float* fcb = (const float*)d_in[15];
  __bf16* wsb = (__bf16*)d_ws;
  const int E = in_sizes[3];

  hipMemsetAsync(d_out, 0, (size_t)out_size * sizeof(float), stream);

  auto prep = [&](const float* s, size_t off, int K, int N) {
    const int tot = K * N;
    wprep_kernel<<<(tot + 255) / 256, 256, 0, stream>>>(s, wsb + off, K, N);
  };
  prep(w0, OFF_W0, 160, 160);
  prep(w1r, OFF_W1R, 128, 128);
  prep(w1i, OFF_W1I, 128, 128);
  prep(w2r, OFF_W2R, 96, 96);
  prep(w2i, OFF_W2I, 96, 96);
  prep(w3r, OFF_W3R, 64, 64);
  prep(w3i, OFF_W3I, 64, 64);
  prep(w4r, OFF_W4R, 32, 32);
  prep(w4i, OFF_W4I, 32, 32);
  prep(fcw, OFF_FCW, 64, 160);

  const int NT = E / 16;  // E = 160000 -> 10000 tiles of 16 edges
  so2_equiconv_main<<<1024, WAVES * 32, 0, stream>>>(x, wig, efc, esrc, edst, fcb, wsb,
                                                     (float*)d_out, NT);
}
